// CausalSelfAttention_20280835572116
// MI455X (gfx1250) — matrix operand order repaired and checked by reading, not compiled, not run on hardware
//
#include <hip/hip_runtime.h>

typedef _Float16 f16;
typedef __attribute__((ext_vector_type(16))) _Float16 v16h;
typedef __attribute__((ext_vector_type(8)))  _Float16 v8h;
typedef __attribute__((ext_vector_type(8)))  float    v8f;

#define Bn 2
#define Tn 2048
#define Dm 1024
#define Hn 16
#define HD 64

// Build a WMMA A-operand (16x32 f16) fragment for this lane.
// k order used here (B loads match): lanes 0-15 hold K={0..15}, lanes 16-31 hold K={16..31}
// for row M = lane%16.  offA = (lane<16) ? 0 : 8.
static __device__ inline v16h make_a(const f16* base, int offA) {
  v8h lo = *(const v8h*)(base + 2 * offA);
  v8h hi = *(const v8h*)(base + 2 * offA + 8);
  v16h a;
#pragma unroll
  for (int i = 0; i < 8; ++i) { a[i] = lo[i]; a[i + 8] = hi[i]; }
  return a;
}

// CDNA5 async global->LDS copy (16B per lane), tracked with ASYNCcnt.
static __device__ inline void async_b128(const f16* g, f16* l) {
  unsigned lds_addr = (unsigned)(unsigned long long)l;   // generic low32 == LDS offset
  asm volatile("global_load_async_to_lds_b128 %0, %1, off"
               :: "v"(lds_addr), "v"((unsigned long long)g)
               : "memory");
}

// DPP16 lane permute (row-confined: never crosses a 16-lane half).
template <int CTRL>
static __device__ inline float dppf(float v) {
  return __int_as_float(__builtin_amdgcn_update_dpp(
      0, __float_as_int(v), CTRL, 0xF, 0xF, true));
}
// Reduce across each 16-lane half with pure VALU DPP (no ds_bpermute / waits):
// quad xor1, quad xor2, half-mirror (joins quads in each 8), row-mirror (joins 8s).
static __device__ inline float half16_max(float x) {
  x = fmaxf(x, dppf<0xB1>(x));    // quad_perm(1,0,3,2)
  x = fmaxf(x, dppf<0x4E>(x));    // quad_perm(2,3,0,1)
  x = fmaxf(x, dppf<0x141>(x));   // ROW_HALF_MIRROR
  x = fmaxf(x, dppf<0x140>(x));   // ROW_MIRROR
  return x;
}
static __device__ inline float half16_sum(float x) {
  x += dppf<0xB1>(x);
  x += dppf<0x4E>(x);
  x += dppf<0x141>(x);
  x += dppf<0x140>(x);
  return x;
}

// ---------------- RMSNorm: x (f32) -> h (f16) ----------------
__global__ void __launch_bounds__(256) rmsnorm_kernel(const float* __restrict__ x,
                                                      const float* __restrict__ w,
                                                      f16* __restrict__ hout) {
  __shared__ float red[256];
  const int row = blockIdx.x;
  const int tid = threadIdx.x;
  const float* xr = x + (size_t)row * Dm;
  float s = 0.f;
  for (int i = tid; i < Dm; i += 256) s += xr[i] * xr[i];
  red[tid] = s;
  __syncthreads();
  for (int off = 128; off > 0; off >>= 1) {
    if (tid < off) red[tid] += red[tid + off];
    __syncthreads();
  }
  const float inv = rsqrtf(red[0] / (float)Dm + 1e-6f);
  for (int i = tid; i < Dm; i += 256)
    hout[(size_t)row * Dm + i] = (f16)(xr[i] * inv * w[i]);
}

// ---------------- f32 -> f16 weight convert ----------------
__global__ void __launch_bounds__(256) cvt_f32_f16(const float* __restrict__ in,
                                                   f16* __restrict__ out, int n) {
  int i = blockIdx.x * blockDim.x + threadIdx.x;
  if (i < n) out[i] = (f16)in[i];
}

// ---------------- WMMA GEMM: C[m][n] = sum_k A[m][k] * W[n][k] ----------------
// Each wave computes a 16x64 tile, register double-buffered over K so global
// loads for step k+32 overlap the WMMAs of step k.  launch_bounds(...,1) gives
// the allocator the full VGPR budget (no spills; latency hidden by pipelining).
template <int N_TOT, bool RES>
__global__ void __launch_bounds__(256, 1)
gemm_wmma(const f16* __restrict__ A, const f16* __restrict__ W,
          const float* __restrict__ resid, void* __restrict__ outp) {
  const int lane = threadIdx.x & 31;
  const int wave = threadIdx.x >> 5;
  const int m0 = blockIdx.y * 128 + wave * 16;
  const int n0 = blockIdx.x * 64;
  const int l15 = lane & 15;
  const int offA = (lane < 16) ? 0 : 8;
  const int offB = (lane < 16) ? 0 : 16;

  const f16* arow = A + (size_t)(m0 + l15) * Dm;
  const f16* brow[4];
#pragma unroll
  for (int tn = 0; tn < 4; ++tn)
    brow[tn] = W + (size_t)(n0 + tn * 16 + l15) * Dm + offB;

  v8f acc[4] = {};
  v16h a_cur = make_a(arow, offA);
  v16h b_cur[4];
#pragma unroll
  for (int tn = 0; tn < 4; ++tn) b_cur[tn] = *(const v16h*)(brow[tn]);

  for (int k0 = 0; k0 < Dm; k0 += 32) {
    const int kp = (k0 + 32 < Dm) ? (k0 + 32) : 0;  // clamped prefetch (uniform)
    v16h a_nxt = make_a(arow + kp, offA);
    v16h b_nxt[4];
#pragma unroll
    for (int tn = 0; tn < 4; ++tn) b_nxt[tn] = *(const v16h*)(brow[tn] + kp);
#pragma unroll
    for (int tn = 0; tn < 4; ++tn)
      acc[tn] = __builtin_amdgcn_wmma_f32_16x16x32_f16(
          false, a_cur, false, b_cur[tn], (short)0, acc[tn], false, false);
    a_cur = a_nxt;
#pragma unroll
    for (int tn = 0; tn < 4; ++tn) b_cur[tn] = b_nxt[tn];
  }

  const int rbase = m0 + ((lane >= 16) ? 8 : 0);
#pragma unroll
  for (int tn = 0; tn < 4; ++tn) {
    const int col = n0 + tn * 16 + l15;
#pragma unroll
    for (int r = 0; r < 8; ++r) {
      const int row = rbase + r;
      if (RES) {
        float* o = (float*)outp;
        o[(size_t)row * N_TOT + col] = resid[(size_t)row * N_TOT + col] + acc[tn][r];
      } else {
        f16* o = (f16*)outp;
        o[(size_t)row * N_TOT + col] = (f16)acc[tn][r];
      }
    }
  }
}

// ---------------- RoPE + repack q,k,v ----------------
__global__ void __launch_bounds__(256) rope_repack(const f16* __restrict__ qkv,
                                                   f16* __restrict__ Qr,
                                                   f16* __restrict__ Kr,
                                                   f16* __restrict__ Vt) {
  const int tid = blockIdx.x * blockDim.x + threadIdx.x;  // B*H*T*32 threads
  const int j = tid & 31;
  const int t = (tid >> 5) & (Tn - 1);
  const int h = (tid >> 16) & (Hn - 1);
  const int b = tid >> 20;
  if (b >= Bn) return;
  const size_t base = ((size_t)(b * Tn + t)) * (3 * Dm) + h * HD;
  const float qj  = (float)qkv[base + j];
  const float qj2 = (float)qkv[base + j + 32];
  const float kj  = (float)qkv[base + Dm + j];
  const float kj2 = (float)qkv[base + Dm + j + 32];
  const f16 vj  = qkv[base + 2 * Dm + j];
  const f16 vj2 = qkv[base + 2 * Dm + j + 32];
  const float inv = powf(10000.0f, -(float)j / 32.0f);
  const float ang = (float)t * inv;
  const float c = cosf(ang), s = sinf(ang);
  const size_t bh = (size_t)(b * Hn + h);
  const size_t qbase = (bh * Tn + t) * HD;
  Qr[qbase + j]      = (f16)(qj * c - qj2 * s);
  Qr[qbase + j + 32] = (f16)(qj2 * c + qj * s);
  Kr[qbase + j]      = (f16)(kj * c - kj2 * s);
  Kr[qbase + j + 32] = (f16)(kj2 * c + kj * s);
  Vt[(bh * HD + j) * Tn + t]      = vj;
  Vt[(bh * HD + j + 32) * Tn + t] = vj2;
}

// ---------------- Flash attention (causal, online softmax) ----------------
// grid (T/64, H, B), block 128 (4 waves).  Each wave: 16 q-rows.
// 32-key K/V tiles staged once per block into double-buffered LDS via
// global_load_async_to_lds_b128 (ASYNCcnt), overlapping DMA with WMMA+softmax.
// launch_bounds(128,1): full VGPR budget so all fragments/stats stay resident.
__global__ void __launch_bounds__(128, 1)
attn_kernel(const f16* __restrict__ Qr, const f16* __restrict__ Kr,
            const f16* __restrict__ Vt, f16* __restrict__ ctx) {
  __shared__ __align__(64) f16 plds[4][16 * 32];  // per-wave P staging
  __shared__ __align__(64) f16 Ks[2][32 * 64];    // [key][d]
  __shared__ __align__(64) f16 Vs[2][64 * 32];    // [d][key]
  const int tid = threadIdx.x;
  const int lane = tid & 31;
  const int wave = tid >> 5;
  const int b = blockIdx.z, h = blockIdx.y;
  const int m0 = blockIdx.x * 64 + wave * 16;
  const size_t bh = (size_t)(b * Hn + h);
  const f16* Qb = Qr + bh * Tn * HD;
  const f16* Kb = Kr + bh * Tn * HD;
  const f16* Vb = Vt + bh * HD * Tn;

  const int l15 = lane & 15;
  const int offA = (lane < 16) ? 0 : 8;
  const int offB = (lane < 16) ? 0 : 16;

  const f16* qrow = Qb + (size_t)(m0 + l15) * HD;
  const v16h aq0 = make_a(qrow, offA);       // d = 0..31
  const v16h aq1 = make_a(qrow + 32, offA);  // d = 32..63

  v8f o[4] = {};
  float mstat[8], lstat[8];
#pragma unroll
  for (int r = 0; r < 8; ++r) { mstat[r] = -1e30f; lstat[r] = 0.f; }

  const int rbase = m0 + ((lane >= 16) ? 8 : 0);
  const int nkt = blockIdx.x * 2 + 2;  // keys [0, m0_block+64): uniform per block
  f16* myp = &plds[wave][0];

  // Stage one 32-key tile: K tile = 32 rows x 128B, V tile = 64 rows x 64B.
  // 4 async b128 instructions per wave per tile (2 K + 2 V).
  auto issue_tile = [&](int kt, int buf) {
    const int n0 = kt * 32;
#pragma unroll
    for (int i = 0; i < 2; ++i) {
      const int c = tid + i * 128;                 // chunk 0..255 (16B chunks)
      const int kr = c >> 3, ksg = c & 7;          // K: row, 8-half segment
      async_b128(Kb + (size_t)(n0 + kr) * HD + ksg * 8, &Ks[buf][c * 8]);
      const int vd = c >> 2, vsg = c & 3;          // V: d-row, segment
      async_b128(Vb + (size_t)vd * Tn + n0 + vsg * 8, &Vs[buf][c * 8]);
    }
  };

  issue_tile(0, 0);  // prologue

  for (int kt = 0; kt < nkt; ++kt) {
    const int cur = kt & 1;
    const int ktn = (kt + 1 < nkt) ? (kt + 1) : kt;  // clamped (uniform)
    issue_tile(ktn, 1 - cur);                        // prefetch next tile
    asm volatile("s_wait_asynccnt 0x4" ::: "memory");  // tile kt landed
    __syncthreads();                                   // visible to all waves

    const int n0 = kt * 32;
    // Hoist all K B-fragments, then issue the 4 S WMMAs.
    v16h kb[2][2];
#pragma unroll
    for (int st = 0; st < 2; ++st) {
      const f16* krow = &Ks[cur][(st * 16 + l15) * HD];
      kb[st][0] = *(const v16h*)(krow + offB);       // d 0..31
      kb[st][1] = *(const v16h*)(krow + 32 + offB);  // d 32..63
    }
    v8f S[2] = {};
#pragma unroll
    for (int st = 0; st < 2; ++st) {
      S[st] = __builtin_amdgcn_wmma_f32_16x16x32_f16(
          false, aq0, false, kb[st][0], (short)0, S[st], false, false);
      S[st] = __builtin_amdgcn_wmma_f32_16x16x32_f16(
          false, aq1, false, kb[st][1], (short)0, S[st], false, false);
    }
    // Issue V B-fragment loads now; their latency hides under the softmax VALU.
    v16h bv[4];
#pragma unroll
    for (int tn = 0; tn < 4; ++tn)
      bv[tn] = *(const v16h*)(&Vs[cur][(tn * 16 + l15) * 32] + offB);

    // Online softmax, row-by-row; rows live in 16-lane halves, reductions are
    // pure-VALU DPP16 (no ds_bpermute, no dscnt waits).
#pragma unroll
    for (int r = 0; r < 8; ++r) {
      const int row = rbase + r;
      float v0 = ((n0 + l15) <= row) ? S[0][r] * 0.125f : -1e30f;
      float v1 = ((n0 + 16 + l15) <= row) ? S[1][r] * 0.125f : -1e30f;
      const float mx = half16_max(fmaxf(v0, v1));
      const float mnew = fmaxf(mstat[r], mx);
      const float sc = __expf(mstat[r] - mnew);
      const float p0 = __expf(v0 - mnew);
      const float p1 = __expf(v1 - mnew);
      const float rs = half16_sum(p0 + p1);
      lstat[r] = lstat[r] * sc + rs;
      mstat[r] = mnew;
#pragma unroll
      for (int tn = 0; tn < 4; ++tn) o[tn][r] *= sc;
      const int prow = r + ((lane >= 16) ? 8 : 0);
      myp[prow * 32 + l15] = (f16)p0;
      myp[prow * 32 + 16 + l15] = (f16)p1;
    }
    // P region is wave-private; wave-local DS ordering suffices (LDS in-order).
    asm volatile("s_wait_dscnt 0x0" ::: "memory");
    const v16h ap = make_a(myp + l15 * 32, offA);  // C-layout -> A-layout
#pragma unroll
    for (int tn = 0; tn < 4; ++tn)
      o[tn] = __builtin_amdgcn_wmma_f32_16x16x32_f16(
          false, ap, false, bv[tn], (short)0, o[tn], false, false);
    __syncthreads();  // all reads of buf `cur` done before next DMA reuses it
  }

  // normalize + write ctx [b][t][h*64+dn] as f16
#pragma unroll
  for (int tn = 0; tn < 4; ++tn) {
#pragma unroll
    for (int r = 0; r < 8; ++r) {
      const int row = rbase + r;
      const int dn = tn * 16 + l15;
      const float val = o[tn][r] / lstat[r];
      ctx[((size_t)(b * Tn + row)) * Dm + h * HD + dn] = (f16)val;
    }
  }
}

extern "C" void kernel_launch(void* const* d_in, const int* in_sizes, int n_in,
                              void* d_out, int out_size, void* d_ws, size_t ws_size,
                              hipStream_t stream) {
  (void)in_sizes; (void)n_in; (void)out_size; (void)ws_size;
  const float* x      = (const float*)d_in[0];
  const float* norm_w = (const float*)d_in[1];
  const float* w_qkv  = (const float*)d_in[2];
  const float* w_out  = (const float*)d_in[3];
  float* out = (float*)d_out;

  char* ws = (char*)d_ws;
  size_t off = 0;
  auto alloc = [&](size_t bytes) -> void* {
    void* p = ws + off;
    off += (bytes + 255) & ~(size_t)255;
    return p;
  };
  f16* hbuf = (f16*)alloc((size_t)Bn * Tn * Dm * 2);          // 8 MB
  f16* wq16 = (f16*)alloc((size_t)3 * Dm * Dm * 2);           // 6 MB
  f16* wo16 = (f16*)alloc((size_t)Dm * Dm * 2);               // 2 MB
  f16* qkv  = (f16*)alloc((size_t)Bn * Tn * 3 * Dm * 2);      // 24 MB
  f16* Qr   = (f16*)alloc((size_t)Bn * Hn * Tn * HD * 2);     // 8 MB
  f16* Kr   = (f16*)alloc((size_t)Bn * Hn * Tn * HD * 2);     // 8 MB
  f16* Vt   = (f16*)alloc((size_t)Bn * Hn * HD * Tn * 2);     // 8 MB
  f16* ctx  = (f16*)alloc((size_t)Bn * Tn * Dm * 2);          // 8 MB

  rmsnorm_kernel<<<Bn * Tn, 256, 0, stream>>>(x, norm_w, hbuf);
  cvt_f32_f16<<<(3 * Dm * Dm + 255) / 256, 256, 0, stream>>>(w_qkv, wq16, 3 * Dm * Dm);
  cvt_f32_f16<<<(Dm * Dm + 255) / 256, 256, 0, stream>>>(w_out, wo16, Dm * Dm);
  gemm_wmma<3 * Dm, false><<<dim3(3 * Dm / 64, Bn * Tn / 128), 256, 0, stream>>>(
      hbuf, wq16, nullptr, qkv);
  rope_repack<<<(Bn * Hn * Tn * 32) / 256, 256, 0, stream>>>(qkv, Qr, Kr, Vt);
  attn_kernel<<<dim3(Tn / 64, Hn, Bn), 128, 0, stream>>>(Qr, Kr, Vt, ctx);
  gemm_wmma<Dm, true><<<dim3(Dm / 64, Bn * Tn / 128), 256, 0, stream>>>(
      ctx, wo16, x, out);
}
